// Propagation_26216480375149
// MI455X (gfx1250) — compile-verified, matrix operand order checked by reading, not run on hardware
//
#include <hip/hip_runtime.h>

// ---------------------------------------------------------------------------
// Fused propagation forward for MI455X (gfx1250, wave32, WMMA).
//   scores = val @ val^T ; edges = softsign(scores)
//   delta_val = edges @ val ; delta_state = edges @ state
//   out[b,i,:] = concat(delta_val[b,i,:], delta_state[b,i])
// Flash-style fusion: edges ([B,N,N] = 268MB) never touches HBM.
// bf16 WMMA (v_wmma_f32_16x16x32_bf16) with f32 accumulation.
//
// Round-2 changes:
//  - f32->bf16 via vector __builtin_convertvector -> packed v_cvt_pk_bf16_f32
//    (was ~5 integer VALU per element).
//  - 4x4-block staging: all LDS staging stores are b64 (was 4x b16 per float4).
//  - LDS strides padded so fragment-load bank-start = 4*odd mod 64:
//    Kv/Qs stride 264 (132 dw == 4 mod 64), KvT/Es stride 72 (36 dw == 4*9).
// ---------------------------------------------------------------------------

#define BB 4
#define NN 4096
#define DD 256
#define MTILE 64   // rows i per block
#define JTILE 64   // j columns per inner iteration
#define NJ (NN / JTILE)

#define LDQ 264    // Qs / Kv row stride (bf16 elems), 264*2B, 132 dw % 64 = 4
#define LDT 72     // KvT / Es row stride (bf16 elems), 36 dw % 64 = 36 = 4*9

typedef __bf16 bf16_t;
typedef __attribute__((ext_vector_type(4)))  bf16_t v4bf;
typedef __attribute__((ext_vector_type(8)))  bf16_t v8bf;
typedef __attribute__((ext_vector_type(16))) bf16_t v16bf;
typedef __attribute__((ext_vector_type(4)))  float  v4f;
typedef __attribute__((ext_vector_type(8)))  float  v8f;

// Load one 16x32 bf16 WMMA operand fragment from a row-major LDS matrix.
// CDNA5 16-bit A/B layout: lanes 0-15 hold row m=lane with K = k0+0..7 and
// k0+16..23; lanes 16-31 hold row m=lane-16 with K = k0+8..15 and k0+24..31.
// Both 8-element runs are contiguous -> two ds_load_b128.
__device__ __forceinline__ v16bf load_frag(const bf16_t* base, int rowbase,
                                           int k0, int ld, int lane) {
    const bf16_t* p = base + (rowbase + (lane & 15)) * ld + k0 + ((lane >> 4) << 3);
    v8bf lo = *(const v8bf*)(p);
    v8bf hi = *(const v8bf*)(p + 16);
    return __builtin_shufflevector(lo, hi, 0, 1, 2, 3, 4, 5, 6, 7,
                                           8, 9, 10, 11, 12, 13, 14, 15);
}

__device__ __forceinline__ v8f wmma_bf16(v16bf a, v16bf b, v8f c) {
    return __builtin_amdgcn_wmma_f32_16x16x32_bf16(
        /*neg_a=*/false, a, /*neg_b=*/false, b,
        /*c_mod=*/(short)0, c, /*reuse_a=*/false, /*reuse_b=*/false);
}

__global__ void __launch_bounds__(256)
propagation_fused_kernel(const float* __restrict__ val,
                         const float* __restrict__ state,
                         float* __restrict__ out) {
    extern __shared__ char smem_raw[];
    bf16_t* Qs  = (bf16_t*)smem_raw;        // [64][LDQ] : Q tile (rows i), bf16
    bf16_t* Kv  = Qs  + MTILE * LDQ;        // [64][LDQ] : K/V tile row-major (j,d)
    bf16_t* KvT = Kv  + JTILE * LDQ;        // [256][LDT]: K/V tile transposed (d,j)
    bf16_t* Es  = KvT + DD * LDT;           // [64][LDT] : softsign(S) tile
    bf16_t* Ss  = Es  + MTILE * LDT;        // [64]      : state slice

    const int tid  = threadIdx.x;
    const int lane = tid & 31;
    const int wid  = tid >> 5;              // 8 waves
    const int mt   = wid & 3;               // wave's 16-row M tile (0..3)
    const int nh   = wid >> 2;              // wave's 128-col D half (0..1)
    const int lrow = lane & 15;
    const int hh   = (lane >> 4) << 3;      // 0 or 8 (half-wave row/col offset)

    const int b  = blockIdx.y;              // batch
    const int i0 = blockIdx.x * MTILE;      // row base

    const float* valB   = val + (size_t)b * NN * DD;
    const float* stateB = state + (size_t)b * NN;

    // ---- Stage Q tile: 4x4 f32 blocks -> bf16 LDS (b64 stores) ----
    #pragma unroll
    for (int it = 0; it < 4; ++it) {
        int g  = it * 256 + tid;            // block id 0..1023
        int j  = (g >> 6) << 2;             // local row 0..60 step 4
        int d  = (g & 63) << 2;             // col 0..252 step 4
        v4f r0 = *(const v4f*)(valB + (size_t)(i0 + j + 0) * DD + d);
        v4f r1 = *(const v4f*)(valB + (size_t)(i0 + j + 1) * DD + d);
        v4f r2 = *(const v4f*)(valB + (size_t)(i0 + j + 2) * DD + d);
        v4f r3 = *(const v4f*)(valB + (size_t)(i0 + j + 3) * DD + d);
        *(v4bf*)(Qs + (j + 0) * LDQ + d) = __builtin_convertvector(r0, v4bf);
        *(v4bf*)(Qs + (j + 1) * LDQ + d) = __builtin_convertvector(r1, v4bf);
        *(v4bf*)(Qs + (j + 2) * LDQ + d) = __builtin_convertvector(r2, v4bf);
        *(v4bf*)(Qs + (j + 3) * LDQ + d) = __builtin_convertvector(r3, v4bf);
    }

    // ---- Persistent accumulators: 16 rows x 128 cols of delta_val per wave ----
    v8f acc[8];
    #pragma unroll
    for (int t = 0; t < 8; ++t) acc[t] = (v8f){};
    v8f sacc = {};                          // delta_state tile (waves 0-3 only)

    const int jt0 = nh * 2;                 // wave's two S j-tiles in GEMM1
    const int jt1 = nh * 2 + 1;

    for (int jiter = 0; jiter < NJ; ++jiter) {
        const int j0 = jiter * JTILE;

        // Wait until the previous iteration's reads of Kv/KvT/Es are done.
        __syncthreads();

        // ---- Stage K/V tile: 4x4 blocks -> Kv (row-major) + KvT (transposed),
        //      all stores are 8-byte ds_store_b64. Loads fully coalesced. ----
        #pragma unroll
        for (int it = 0; it < 4; ++it) {
            int g  = it * 256 + tid;
            int j  = (g >> 6) << 2;
            int d  = (g & 63) << 2;
            const float* src = valB + (size_t)(j0 + j) * DD + d;
            v4f r0 = *(const v4f*)(src);
            v4f r1 = *(const v4f*)(src + DD);
            v4f r2 = *(const v4f*)(src + 2 * DD);
            v4f r3 = *(const v4f*)(src + 3 * DD);
            v4bf c0 = __builtin_convertvector(r0, v4bf);
            v4bf c1 = __builtin_convertvector(r1, v4bf);
            v4bf c2 = __builtin_convertvector(r2, v4bf);
            v4bf c3 = __builtin_convertvector(r3, v4bf);
            *(v4bf*)(Kv + (j + 0) * LDQ + d) = c0;
            *(v4bf*)(Kv + (j + 1) * LDQ + d) = c1;
            *(v4bf*)(Kv + (j + 2) * LDQ + d) = c2;
            *(v4bf*)(Kv + (j + 3) * LDQ + d) = c3;
            #pragma unroll
            for (int c = 0; c < 4; ++c) {
                v4bf tcol = { c0[c], c1[c], c2[c], c3[c] };
                *(v4bf*)(KvT + (d + c) * LDT + j) = tcol;
            }
            // Prefetch next j-tile, one request per 128B cacheline.
            if (((g & 7) == 0) && (jiter + 1 < NJ)) {
                __builtin_prefetch(src + (size_t)JTILE * DD, 0, 3);
                __builtin_prefetch(src + (size_t)(JTILE + 1) * DD, 0, 3);
                __builtin_prefetch(src + (size_t)(JTILE + 2) * DD, 0, 3);
                __builtin_prefetch(src + (size_t)(JTILE + 3) * DD, 0, 3);
            }
        }
        if (tid < JTILE)
            Ss[tid] = (bf16_t)stateB[j0 + tid];
        __syncthreads();

        // ---- GEMM1: S[64x64] = Q[64x256] @ K[64x256]^T, 2 tiles per wave ----
        v8f s0 = {}, s1 = {};
        #pragma unroll
        for (int kc = 0; kc < 8; ++kc) {
            v16bf a  = load_frag(Qs, mt  * 16, kc * 32, LDQ, lane);
            v16bf b0 = load_frag(Kv, jt0 * 16, kc * 32, LDQ, lane);
            v16bf b1 = load_frag(Kv, jt1 * 16, kc * 32, LDQ, lane);
            s0 = wmma_bf16(a, b0, s0);
            s1 = wmma_bf16(a, b1, s1);
        }

        // ---- softsign + convert, write E tiles to LDS (C-layout -> row-major) ----
        #pragma unroll
        for (int r = 0; r < 8; ++r) {
            int m = mt * 16 + r + hh;
            float x0 = s0[r];
            float x1 = s1[r];
            Es[m * LDT + jt0 * 16 + lrow] = (bf16_t)(x0 / (1.0f + __builtin_fabsf(x0)));
            Es[m * LDT + jt1 * 16 + lrow] = (bf16_t)(x1 / (1.0f + __builtin_fabsf(x1)));
        }
        __syncthreads();

        // ---- GEMM2: acc[64x256] += E[64x64] @ V[64x256]; state folded in ----
        #pragma unroll
        for (int kc2 = 0; kc2 < 2; ++kc2) {
            v16bf a = load_frag(Es, mt * 16, kc2 * 32, LDT, lane);
            #pragma unroll
            for (int t = 0; t < 8; ++t) {
                v16bf bv = load_frag(KvT, (nh * 8 + t) * 16, kc2 * 32, LDT, lane);
                acc[t] = wmma_bf16(a, bv, acc[t]);
            }
            if (nh == 0) {                  // wave-uniform: EXEC all-1s inside
                v16bf bs = {};
                if (lrow == 0) {            // only column n==0 carries state
                    int kb = kc2 * 32 + hh;
                    #pragma unroll
                    for (int t = 0; t < 8; ++t) {
                        bs[t]     = Ss[kb + t];
                        bs[t + 8] = Ss[kb + 16 + t];
                    }
                }
                sacc = wmma_bf16(a, bs, sacc);
            }
        }
    }

    // ---- Writeback: out[b,i,0:256] = delta_val, out[b,i,256] = delta_state ----
    #pragma unroll
    for (int t = 0; t < 8; ++t) {
        int d = nh * 128 + t * 16 + lrow;
        #pragma unroll
        for (int r = 0; r < 8; ++r) {
            size_t irow = (size_t)b * NN + i0 + mt * 16 + r + hh;
            out[irow * (DD + 1) + d] = acc[t][r];
        }
    }
    if (nh == 0 && lrow == 0) {
        #pragma unroll
        for (int r = 0; r < 8; ++r) {
            size_t irow = (size_t)b * NN + i0 + mt * 16 + r + hh;
            out[irow * (DD + 1) + DD] = sacc[r];
        }
    }
}

extern "C" void kernel_launch(void* const* d_in, const int* in_sizes, int n_in,
                              void* d_out, int out_size, void* d_ws, size_t ws_size,
                              hipStream_t stream) {
    (void)in_sizes; (void)n_in; (void)out_size; (void)d_ws; (void)ws_size;
    const float* val   = (const float*)d_in[0];   // [B,N,D] f32
    const float* state = (const float*)d_in[1];   // [B,N]   f32
    float* out = (float*)d_out;                   // [B,N,D+1] f32

    dim3 grid(NN / MTILE, BB);                    // 64 x 4 blocks
    dim3 block(256);                              // 8 wave32 waves
    size_t smem = (size_t)(MTILE * LDQ +          // Qs
                           JTILE * LDQ +          // Kv
                           DD * LDT +             // KvT
                           MTILE * LDT +          // Es
                           JTILE)                 // Ss
                  * sizeof(bf16_t);               // ~112 KB
    propagation_fused_kernel<<<grid, block, smem, stream>>>(val, state, out);
}
